// MoE_76450417869448
// MI455X (gfx1250) — matrix operand order repaired and checked by reading, not compiled, not run on hardware
//
#include <hip/hip_runtime.h>
#include <hip/hip_bf16.h>

typedef __bf16 bf16;
typedef __attribute__((ext_vector_type(16))) __bf16 v16bf;
typedef __attribute__((ext_vector_type(8)))  __bf16 v8bf;
typedef __attribute__((ext_vector_type(8)))  float   v8f;

#define NB 32      // batch
#define NS 2048    // sequence
#define KD 1024    // in dim (K)
#define ND 2048    // out dim (N)
#define NE 16      // experts

#define MT 128     // block tile M
#define NT 256     // block tile N
#define KT 32      // K stage (one bf16 WMMA K)
#define LDSTR 40   // padded LDS row stride in bf16 elems (80B, 16B aligned, bank-friendly)

__device__ __forceinline__ int top1_expert(const float* __restrict__ probs, int b) {
    float best = probs[b * NE];
    int e = 0;
#pragma unroll
    for (int i = 1; i < NE; ++i) {
        float p = probs[b * NE + i];
        if (p > best) { best = p; e = i; }
    }
    return e;
}

__device__ __forceinline__ v16bf combine8(v8bf lo, v8bf hi) {
    return __builtin_shufflevector(lo, hi, 0,1,2,3,4,5,6,7,8,9,10,11,12,13,14,15);
}

// global fp32 tile -> registers (A: 128x32 = 4 float4/thread, B: 256x32 = 8 float4/thread)
__device__ __forceinline__ void stage_load(const float* __restrict__ xb,
                                           const float* __restrict__ we,
                                           int k0, int tid,
                                           float4* aReg, float4* bReg) {
#pragma unroll
    for (int i = 0; i < 4; ++i) {
        int id  = tid + i * 256;
        aReg[i] = *(const float4*)(xb + (size_t)(id >> 3) * KD + k0 + (id & 7) * 4);
    }
#pragma unroll
    for (int i = 0; i < 8; ++i) {
        int id  = tid + i * 256;
        bReg[i] = *(const float4*)(we + (size_t)(id >> 3) * KD + k0 + (id & 7) * 4);
    }
}

// fp32 regs -> bf16 hi/lo split -> LDS tiles
__device__ __forceinline__ void stage_store(bf16* __restrict__ aHi, bf16* __restrict__ aLo,
                                            bf16* __restrict__ bHi, bf16* __restrict__ bLo,
                                            const float4* aReg, const float4* bReg, int tid) {
#pragma unroll
    for (int i = 0; i < 4; ++i) {
        int id  = tid + i * 256;
        int row = id >> 3;
        int c4  = id & 7;
        float va[4] = {aReg[i].x, aReg[i].y, aReg[i].z, aReg[i].w};
#pragma unroll
        for (int j = 0; j < 4; ++j) {
            float f  = va[j];
            float fh = __uint_as_float(__float_as_uint(f) & 0xffff0000u);
            aHi[row * LDSTR + c4 * 4 + j] = (bf16)fh;
            aLo[row * LDSTR + c4 * 4 + j] = (bf16)(f - fh);
        }
    }
#pragma unroll
    for (int i = 0; i < 8; ++i) {
        int id  = tid + i * 256;
        int row = id >> 3;
        int c4  = id & 7;
        float vb[4] = {bReg[i].x, bReg[i].y, bReg[i].z, bReg[i].w};
#pragma unroll
        for (int j = 0; j < 4; ++j) {
            float g  = vb[j];
            float gh = __uint_as_float(__float_as_uint(g) & 0xffff0000u);
            bHi[row * LDSTR + c4 * 4 + j] = (bf16)gh;
            bLo[row * LDSTR + c4 * 4 + j] = (bf16)(g - gh);
        }
    }
}

__global__ __launch_bounds__(256)
void moe_gemm_kernel(const float* __restrict__ x,
                     const float* __restrict__ probs,
                     const float* __restrict__ w,
                     const float* __restrict__ bias,
                     float* __restrict__ out) {
    // double-buffered bf16 hi/lo tiles: 2 * (10+10+20+20) KB = 120 KB (<320 KB WGP LDS)
    __shared__ __align__(16) bf16 sAhi[2][MT * LDSTR];
    __shared__ __align__(16) bf16 sAlo[2][MT * LDSTR];
    __shared__ __align__(16) bf16 sBhi[2][NT * LDSTR];
    __shared__ __align__(16) bf16 sBlo[2][NT * LDSTR];

    const int b     = blockIdx.z;
    const int baseS = blockIdx.x * MT;
    const int baseO = blockIdx.y * NT;
    const int e     = top1_expert(probs, b);

    const float* __restrict__ xb = x + (size_t)b * NS * KD + (size_t)baseS * KD;
    const float* __restrict__ we = w + (size_t)e * ND * KD + (size_t)baseO * KD;

    const int tid  = threadIdx.x;
    const int lane = tid & 31;
    const int wv   = tid >> 5;   // 8 waves
    const int wm   = wv >> 2;    // 0..1 : wave rows 64*wm
    const int wn   = wv & 3;     // 0..3 : wave cols 64*wn
    const int lh   = lane & 15;
    const int lsel = lane >> 4;  // half-wave select

    v8f acc[4][4] = {};          // 64x64 wave tile = 4x4 16x16 f32 accumulators

    float4 aReg[4], bReg[8];

    // prologue: stage 0 -> regs -> LDS buf 0
    stage_load(xb, we, 0, tid, aReg, bReg);
    stage_store(sAhi[0], sAlo[0], sBhi[0], sBlo[0], aReg, bReg, tid);
    __syncthreads();

    const int KSTAGES = KD / KT;  // 32
    for (int ks = 0; ks < KSTAGES; ++ks) {
        const int cur = ks & 1;
        const int nxt = cur ^ 1;

        // issue next stage's global loads before compute (hidden under WMMAs)
        if (ks + 1 < KSTAGES) {
            stage_load(xb, we, (ks + 1) * KT, tid, aReg, bReg);
        }
        // L2 prefetch two stages ahead: A: one per 64B, B: one per 128B row segment
        if (ks + 2 < KSTAGES) {
            int k2 = (ks + 2) * KT;
            __builtin_prefetch(xb + (size_t)(tid >> 1) * KD + k2 + (tid & 1) * 16, 0, 1);
            __builtin_prefetch(we + (size_t)tid * KD + k2, 0, 1);
        }

        // fragments in contiguous K order (k = 16*lsel + s), same for A and B
        const int kbA = lsel * 16;   // A: K run   [kb..kb+15]
        const int kbB = lsel * 16;   // B: K run   [kb..kb+15]

        const bf16* __restrict__ cAhi = sAhi[cur];
        const bf16* __restrict__ cAlo = sAlo[cur];
        const bf16* __restrict__ cBhi = sBhi[cur];
        const bf16* __restrict__ cBlo = sBlo[cur];

        // all 4 B hi/lo fragment pairs resident; stream A per M-subtile
        v16bf bh[4], bl[4];
#pragma unroll
        for (int ni = 0; ni < 4; ++ni) {
            const bf16* pbh = &cBhi[(wn * 64 + ni * 16 + lh) * LDSTR + kbB];
            const bf16* pbl = &cBlo[(wn * 64 + ni * 16 + lh) * LDSTR + kbB];
            bh[ni] = combine8(*(const v8bf*)pbh, *(const v8bf*)(pbh + 8));
            bl[ni] = combine8(*(const v8bf*)pbl, *(const v8bf*)(pbl + 8));
        }
#pragma unroll
        for (int mi = 0; mi < 4; ++mi) {
            const bf16* pah = &cAhi[(wm * 64 + mi * 16 + lh) * LDSTR + kbA];
            const bf16* pal = &cAlo[(wm * 64 + mi * 16 + lh) * LDSTR + kbA];
            v16bf ah = combine8(*(const v8bf*)pah, *(const v8bf*)(pah + 8));
            v16bf al = combine8(*(const v8bf*)pal, *(const v8bf*)(pal + 8));
            // bf16x3: hi*hi + hi*lo + lo*hi, f32 accumulate
#pragma unroll
            for (int ni = 0; ni < 4; ++ni) {
                acc[mi][ni] = __builtin_amdgcn_wmma_f32_16x16x32_bf16(
                    false, ah, false, bh[ni], (short)0, acc[mi][ni], false, false);
                acc[mi][ni] = __builtin_amdgcn_wmma_f32_16x16x32_bf16(
                    false, ah, false, bl[ni], (short)0, acc[mi][ni], false, false);
                acc[mi][ni] = __builtin_amdgcn_wmma_f32_16x16x32_bf16(
                    false, al, false, bh[ni], (short)0, acc[mi][ni], false, false);
            }
        }

        // store next stage into the other buffer (overlaps with this stage's WMMAs;
        // safe: buffer 'nxt' was fully consumed in stage ks-1, fenced by last barrier)
        if (ks + 1 < KSTAGES) {
            stage_store(sAhi[nxt], sAlo[nxt], sBhi[nxt], sBlo[nxt], aReg, bReg, tid);
        }
        __syncthreads();
    }

    // epilogue: add bias, store C (v8f element r -> M = r + 8*lsel, N = lh)
#pragma unroll
    for (int mi = 0; mi < 4; ++mi) {
#pragma unroll
        for (int ni = 0; ni < 4; ++ni) {
            int N  = wn * 64 + ni * 16 + lh;
            float bv = bias[e * ND + baseO + N];
#pragma unroll
            for (int r = 0; r < 8; ++r) {
                int M = wm * 64 + mi * 16 + r + 8 * lsel;
                out[((size_t)b * NS + baseS + M) * ND + (baseO + N)] = acc[mi][ni][r] + bv;
            }
        }
    }
}

__global__ void moe_idx_kernel(const float* __restrict__ probs, float* __restrict__ outIdx) {
    int b = threadIdx.x;
    if (b < NB) outIdx[b] = (float)top1_expert(probs, b);
}

extern "C" void kernel_launch(void* const* d_in, const int* in_sizes, int n_in,
                              void* d_out, int out_size, void* d_ws, size_t ws_size,
                              hipStream_t stream) {
    (void)in_sizes; (void)n_in; (void)out_size; (void)d_ws; (void)ws_size;
    const float* x     = (const float*)d_in[0];
    const float* probs = (const float*)d_in[1];
    const float* w     = (const float*)d_in[2];
    const float* bias  = (const float*)d_in[3];
    float* out = (float*)d_out;

    dim3 grid(NS / MT, ND / NT, NB);
    moe_gemm_kernel<<<grid, 256, 0, stream>>>(x, probs, w, bias, out);
    moe_idx_kernel<<<1, NB, 0, stream>>>(probs, out + (size_t)NB * NS * ND);
}
